// HashRoutedSSMLayer_16793322127760
// MI455X (gfx1250) — matrix operand order repaired and checked by reading, not compiled, not run on hardware
//
#include <hip/hip_runtime.h>
#include <hip/hip_bf16.h>

typedef __attribute__((ext_vector_type(16))) __bf16 v16bf;
typedef __attribute__((ext_vector_type(8)))  float  v8f;
typedef __attribute__((ext_vector_type(4)))  unsigned int u32x4;
typedef __attribute__((ext_vector_type(8)))  int i32x8;
typedef __attribute__((ext_vector_type(4)))  int i32x4;

#define NB_ 8
#define TLEN 2048
#define NTOK (NB_ * TLEN)          // 16384 tokens
#define DIM 1024
#define STATE 128
#define SELH 256
#define EXPERTS 8
#define NTILES_MAX ((NTOK / 16) + EXPERTS)   // 1032 padded tiles max
#define BUCKET_LEN (NTOK + 16 * EXPERTS)

// ---------- helpers ----------
__device__ __forceinline__ __bf16 f2bf(float f) {
  union { float f; unsigned u; } a; a.f = f;
  unsigned u = a.u;
  u += 0x7FFFu + ((u >> 16) & 1u);               // round-to-nearest-even
  union { unsigned short s; __bf16 b; } c; c.s = (unsigned short)(u >> 16);
  return c.b;
}
__device__ __forceinline__ float sigmoidf_(float x) { return 1.f / (1.f + __expf(-x)); }

// A fragment (16x32 bf16) from an LDS row: two contiguous 8-half runs
__device__ __forceinline__ v16bf load_afrag(const __bf16* row, int kbase, int aKofs) {
  v16bf af;
  const __bf16* p0 = row + kbase + aKofs;
#pragma unroll
  for (int j = 0; j < 8; ++j) af[j] = p0[j];
  const __bf16* p1 = row + kbase + 16 + aKofs;
#pragma unroll
  for (int j = 0; j < 8; ++j) af[8 + j] = p1[j];
  return af;
}

// meta layout (ints): [0..7]=counts  [8..15]=cursors  [16..24]=padded offsets (9)
__global__ void k_meta_init(int* meta) {
  int i = threadIdx.x;
  if (i < 64) meta[i] = 0;
}

__global__ void k_bucket_fill(int* bucket, int n) {
  int i = blockIdx.x * 256 + threadIdx.x;
  if (i < n) bucket[i] = -1;
}

__global__ void k_routes(const int* __restrict__ tok, int* __restrict__ routes,
                         int* __restrict__ meta, int n) {
  int i = blockIdx.x * 256 + threadIdx.x;
  if (i >= n) return;
  unsigned x = (unsigned)tok[i];
  x ^= x >> 16; x *= 2246822507u;
  x ^= x >> 13; x *= 3266489909u;
  x ^= x >> 16;
  int r = (int)(x % (unsigned)EXPERTS);
  routes[i] = r;
  atomicAdd(&meta[r], 1);
}

__global__ void k_offsets(int* meta) {
  if (threadIdx.x == 0 && blockIdx.x == 0) {
    int off = 0;
    meta[16] = 0;
    for (int e = 0; e < EXPERTS; ++e) {
      int c = meta[e];
      off += (c + 15) & ~15;                    // pad each expert region to 16
      meta[16 + e + 1] = off;
    }
  }
}

__global__ void k_scatter(const int* __restrict__ routes, int* __restrict__ meta,
                          int* __restrict__ bucket, int n) {
  int i = blockIdx.x * 256 + threadIdx.x;
  if (i >= n) return;
  int r = routes[i];
  int pos = atomicAdd(&meta[8 + r], 1);
  bucket[meta[16 + r] + pos] = i;
}

// x (f32, [NTOK][DIM]) -> bf16 image for TDM gather
__global__ void k_cvt_x(const float* __restrict__ src, void* dst_v, int n) {
  __bf16* dst = (__bf16*)dst_v;
  int i = blockIdx.x * 256 + threadIdx.x;
  if (i < n) dst[i] = f2bf(src[i]);
}

// Re-pack f32 weight [E][K][N] -> bf16 WMMA B-fragment-major layout:
//   [(e*NT + nt)*KB + kb][lane(32)][j(16)]
//   lane L, j: holds W[kb*32 + (j&7) + 8*(L>=16) + 16*(j>=8)][nt*16 + (L&15)]
__global__ void k_pack_w(const float* __restrict__ src, void* dst_v, int K, int N) {
  __bf16* dst = (__bf16*)dst_v;
  int total = EXPERTS * K * N;
  int i = blockIdx.x * 256 + threadIdx.x;
  if (i >= total) return;
  int j = i & 15;
  int L = (i >> 4) & 31;
  int rest = i >> 9;                     // (e*NT + nt)*KB + kb
  int KB = K >> 5;
  int kb = rest % KB;
  int ent = rest / KB;
  int NT = N >> 4;
  int nt = ent % NT;
  int e = ent / NT;
  int k = kb * 32 + (j & 7) + ((L >> 4) << 3) + ((j >> 3) << 4);
  int n = nt * 16 + (L & 15);
  dst[i] = f2bf(src[((size_t)e * K + k) * N + n]);
}

// ---------- phase 1: routed GEMMs + gates ----------
__global__ __launch_bounds__(128)
void k_phase1(const void* xbf_v,      // bf16 [NTOK][DIM] image for TDM
              const void* WinP_v,     // packed bf16, K=DIM,  N=STATE
              const void* WsinP_v,    // packed bf16, K=DIM,  N=SELH
              const void* WsoP_v,     // packed bf16, K=SELH, N=4*STATE
              const float* __restrict__ dpar,   // [E][STATE]
              const int* __restrict__ bucket,
              const int* __restrict__ meta,
              float* __restrict__ a_g, float* __restrict__ bu_g,
              float* __restrict__ c_g, float* __restrict__ skip_g) {
  const __bf16* WinP  = (const __bf16*)WinP_v;
  const __bf16* WsinP = (const __bf16*)WsinP_v;
  const __bf16* WsoP  = (const __bf16*)WsoP_v;

  __shared__ __bf16 sA[16][DIM];          // 32 KB (TDM gather destination)
  __shared__ __bf16 sSel[16][SELH];       // 8 KB
  __shared__ float  sU[16][STATE];        // 8 KB
  __shared__ float  sAbcd[16][4 * STATE]; // 32 KB
  __shared__ int    toks[16];
  __shared__ int    sE;

  int tile = blockIdx.x;
  int tid = threadIdx.x;
  int total = meta[24];
  if (tile * 16 >= total) return;

  if (tid == 0) {
    int e = 0;
    for (int k = 1; k < EXPERTS; ++k)
      if (tile * 16 >= meta[16 + k]) e = k;
    sE = e;
  }
  if (tid < 16) toks[tid] = bucket[tile * 16 + tid];
  __syncthreads();
  int e = sE;

  const int wave = tid >> 5;
  const int lane = tid & 31;
  const int m = lane & 15;
  const int hi = lane >> 4;
  const int aKofs = hi ? 8 : 0;
  const int laneOff = lane << 4;          // lane*16 halfs within a B fragment

  // ---- TDM gather: 16 token rows of x_bf16 -> sA (one DMA, wave 0 only) ----
  if (wave == 0) {
    unsigned ldsoff = (unsigned)(size_t)&sA[0][0];            // LDS byte offset
    unsigned long long ga = (unsigned long long)(size_t)xbf_v; // global base
    u32x4 g0;
    g0[0] = 0x80000001u;                          // count=1 | gather_mode, 16-bit idx
    g0[1] = ldsoff;                               // lds_addr
    g0[2] = (unsigned)(ga & 0xFFFFFFFFu);         // global_addr[31:0]
    g0[3] = (unsigned)((ga >> 32) & 0x01FFFFFFu)  // global_addr[56:32]
          | 0x80000000u;                          // type=2 ("image")
    i32x8 g1;
    g1[0] = 0x00010000;                           // data_size=1 (2 bytes)
    g1[1] = (int)(1024u << 16);                   // tensor_dim0 = 1024 (lo16)
    g1[2] = (int)(16384u << 16);                  // dim0 hi=0 | tensor_dim1 lo16 = 16384
    g1[3] = (int)(1024u << 16);                   // dim1 hi=0 | tile_dim0 = 1024
    g1[4] = 16;                                   // tile_dim1 = 16 gather indices
    g1[5] = 1024;                                 // tensor_dim0_stride = 1024
    g1[6] = 0;
    g1[7] = 0;
    int idx[16];
#pragma unroll
    for (int r = 0; r < 16; ++r) { int t = toks[r]; idx[r] = (t < 0) ? 0 : t; }
    i32x4 g2, g3;
#pragma unroll
    for (int r = 0; r < 4; ++r) {
      g2[r] = (idx[2 * r] & 0xFFFF) | (idx[2 * r + 1] << 16);
      g3[r] = (idx[8 + 2 * r] & 0xFFFF) | (idx[8 + 2 * r + 1] << 16);
    }
    i32x8 gz = {0, 0, 0, 0, 0, 0, 0, 0};          // unused extra group (zero-filled)
    __builtin_amdgcn_tensor_load_to_lds(g0, g1, g2, g3, gz, 0);
    __builtin_amdgcn_s_wait_tensorcnt(0);
  }
  __syncthreads();

  // ---- u = X * W_in[e]  (16x1024 @ 1024x128): 8 N-tiles, 2 per wave ----
  {
    const int NT = STATE / 16, KB = DIM / 32;
    int nt0 = wave * 2;
    v8f acc[2];
#pragma unroll
    for (int g = 0; g < 2; ++g) acc[g] = (v8f){};
    for (int kb = 0; kb < KB; ++kb) {
      v16bf af = load_afrag(&sA[m][0], kb * 32, aKofs);
#pragma unroll
      for (int g = 0; g < 2; ++g) {
        const __bf16* bp = WinP + ((((size_t)(e * NT + nt0 + g)) * KB + kb) << 9) + laneOff;
        v16bf bf = *(const v16bf*)bp;
        acc[g] = __builtin_amdgcn_wmma_f32_16x16x32_bf16(false, af, false, bf,
                                                         (short)0, acc[g], false, false);
      }
    }
#pragma unroll
    for (int g = 0; g < 2; ++g)
#pragma unroll
      for (int r = 0; r < 8; ++r) sU[r + 8 * hi][(nt0 + g) * 16 + m] = acc[g][r];
  }

  // ---- sel_h = silu(X * W_sel_in[e]) (1024x256): 16 N-tiles, 4 per wave ----
  {
    const int NT = SELH / 16, KB = DIM / 32;
    int nt0 = wave * 4;
    v8f acc[4];
#pragma unroll
    for (int g = 0; g < 4; ++g) acc[g] = (v8f){};
    for (int kb = 0; kb < KB; ++kb) {
      v16bf af = load_afrag(&sA[m][0], kb * 32, aKofs);
#pragma unroll
      for (int g = 0; g < 4; ++g) {
        const __bf16* bp = WsinP + ((((size_t)(e * NT + nt0 + g)) * KB + kb) << 9) + laneOff;
        v16bf bf = *(const v16bf*)bp;
        acc[g] = __builtin_amdgcn_wmma_f32_16x16x32_bf16(false, af, false, bf,
                                                         (short)0, acc[g], false, false);
      }
    }
#pragma unroll
    for (int g = 0; g < 4; ++g)
#pragma unroll
      for (int r = 0; r < 8; ++r) {
        float v = acc[g][r];
        sSel[r + 8 * hi][(nt0 + g) * 16 + m] = f2bf(v * sigmoidf_(v));
      }
  }
  __syncthreads();

  // ---- abcd = sel_h * W_sel_out[e] (16x256 @ 256x512): 32 N-tiles, 8 per wave ----
  {
    const int NT = (4 * STATE) / 16, KB = SELH / 32;
    for (int gq = 0; gq < 2; ++gq) {
      int nt0 = wave * 8 + gq * 4;
      v8f acc[4];
#pragma unroll
      for (int g = 0; g < 4; ++g) acc[g] = (v8f){};
      for (int kb = 0; kb < KB; ++kb) {
        v16bf af = load_afrag(&sSel[m][0], kb * 32, aKofs);
#pragma unroll
        for (int g = 0; g < 4; ++g) {
          const __bf16* bp = WsoP + ((((size_t)(e * NT + nt0 + g)) * KB + kb) << 9) + laneOff;
          v16bf bf = *(const v16bf*)bp;
          acc[g] = __builtin_amdgcn_wmma_f32_16x16x32_bf16(false, af, false, bf,
                                                           (short)0, acc[g], false, false);
        }
      }
#pragma unroll
      for (int g = 0; g < 4; ++g)
#pragma unroll
        for (int r = 0; r < 8; ++r) sAbcd[r + 8 * hi][(nt0 + g) * 16 + m] = acc[g][r];
    }
  }
  __syncthreads();

  // gates -> per-token streams
  for (int idx = tid; idx < 16 * STATE; idx += 128) {
    int r = idx >> 7;
    int s = idx & (STATE - 1);
    int tk = toks[r];
    if (tk < 0) continue;
    float av = sAbcd[r][s];
    float bv = sAbcd[r][STATE + s];
    float cv = sAbcd[r][2 * STATE + s];
    float dv = sAbcd[r][3 * STATE + s];
    float uu = sU[r][s];
    size_t o = (size_t)tk * STATE + s;
    a_g[o]    = sigmoidf_(av);
    bu_g[o]   = tanhf(bv) * uu;
    c_g[o]    = tanhf(cv);
    skip_g[o] = dpar[e * STATE + s] * sigmoidf_(dv) * uu;
  }
}

// ---------- sequential scan over T: lane = (batch, state channel) ----------
__global__ __launch_bounds__(128)
void k_scan(const int* __restrict__ routes,
            const float* __restrict__ a_g, const float* __restrict__ bu_g,
            float* __restrict__ hseq) {
  int b = blockIdx.x;
  int s = threadIdx.x;
  float h0 = 0.f, h1 = 0.f, h2 = 0.f, h3 = 0.f, h4 = 0.f, h5 = 0.f, h6 = 0.f, h7 = 0.f;
  size_t base = (size_t)b * TLEN;
  for (int t = 0; t < TLEN; ++t) {
    size_t tk = base + t;
    int r = routes[tk];                       // wave-uniform
    size_t o = tk * STATE + s;
    float av = a_g[o];
    float bv = bu_g[o];
    float hp = (r == 0) ? h0 : (r == 1) ? h1 : (r == 2) ? h2 : (r == 3) ? h3
             : (r == 4) ? h4 : (r == 5) ? h5 : (r == 6) ? h6 : h7;
    float hv = av * hp + bv;
    if      (r == 0) h0 = hv;
    else if (r == 1) h1 = hv;
    else if (r == 2) h2 = hv;
    else if (r == 3) h3 = hv;
    else if (r == 4) h4 = hv;
    else if (r == 5) h5 = hv;
    else if (r == 6) h6 = hv;
    else             h7 = hv;
    hseq[o] = hv;
  }
}

// ---------- phase 2: y = c*h + skip; out = y * W_out[e] ----------
__global__ __launch_bounds__(128)
void k_phase2(const float* __restrict__ c_g, const float* __restrict__ skip_g,
              const float* __restrict__ hseq,
              const void* WoutP_v,           // packed bf16, K=STATE, N=DIM
              const int* __restrict__ bucket, const int* __restrict__ meta,
              float* __restrict__ out) {
  const __bf16* WoutP = (const __bf16*)WoutP_v;
  __shared__ __bf16 sY[16][STATE];
  __shared__ int toks[16];
  __shared__ int sE;

  int tile = blockIdx.x;
  int tid = threadIdx.x;
  int total = meta[24];
  if (tile * 16 >= total) return;

  if (tid == 0) {
    int e = 0;
    for (int k = 1; k < EXPERTS; ++k)
      if (tile * 16 >= meta[16 + k]) e = k;
    sE = e;
  }
  if (tid < 16) toks[tid] = bucket[tile * 16 + tid];
  __syncthreads();
  int e = sE;

  for (int idx = tid; idx < 16 * STATE; idx += 128) {
    int r = idx >> 7;
    int s = idx & (STATE - 1);
    int tk = toks[r];
    float y = 0.f;
    if (tk >= 0) {
      size_t o = (size_t)tk * STATE + s;
      y = c_g[o] * hseq[o] + skip_g[o];
    }
    sY[r][s] = f2bf(y);
  }
  __syncthreads();

  const int wave = tid >> 5;
  const int lane = tid & 31;
  const int m = lane & 15;
  const int hi = lane >> 4;
  const int aKofs = hi ? 8 : 0;
  const int laneOff = lane << 4;

  // out tile 16x1024: 64 N-tiles; per wave 16, in 4 groups of 4; K=128 -> 4 kb
  {
    const int NT = DIM / 16, KB = STATE / 32;
    for (int gq = 0; gq < 4; ++gq) {
      int nt0 = wave * 16 + gq * 4;
      v8f acc[4];
#pragma unroll
      for (int g = 0; g < 4; ++g) acc[g] = (v8f){};
#pragma unroll
      for (int kb = 0; kb < KB; ++kb) {
        v16bf af = load_afrag(&sY[m][0], kb * 32, aKofs);
#pragma unroll
        for (int g = 0; g < 4; ++g) {
          const __bf16* bp = WoutP + ((((size_t)(e * NT + nt0 + g)) * KB + kb) << 9) + laneOff;
          v16bf bf = *(const v16bf*)bp;
          acc[g] = __builtin_amdgcn_wmma_f32_16x16x32_bf16(false, af, false, bf,
                                                           (short)0, acc[g], false, false);
        }
      }
#pragma unroll
      for (int g = 0; g < 4; ++g) {
        int n = (nt0 + g) * 16 + m;
#pragma unroll
        for (int r = 0; r < 8; ++r) {
          int mr = r + 8 * hi;
          int tk = toks[mr];
          if (tk >= 0) out[(size_t)tk * DIM + n] = acc[g][r];
        }
      }
    }
  }
}

// ---------- host ----------
extern "C" void kernel_launch(void* const* d_in, const int* in_sizes, int n_in,
                              void* d_out, int out_size, void* d_ws, size_t ws_size,
                              hipStream_t stream) {
  const float* x        = (const float*)d_in[0];
  const int*   tok      = (const int*)d_in[1];
  const float* W_in     = (const float*)d_in[2];
  const float* W_sel_in = (const float*)d_in[3];
  const float* W_sel_out= (const float*)d_in[4];
  const float* W_out    = (const float*)d_in[5];
  const float* d_param  = (const float*)d_in[6];
  float* out = (float*)d_out;

  char* ws = (char*)d_ws;
  size_t off = 0;
  auto alloc = [&](size_t bytes) -> void* {
    void* p = ws + off;
    off = (off + bytes + 255) & ~(size_t)255;
    return p;
  };

  int* meta    = (int*)alloc(64 * sizeof(int));
  int* routes  = (int*)alloc((size_t)NTOK * sizeof(int));
  int* bucket  = (int*)alloc((size_t)BUCKET_LEN * sizeof(int));
  void* xbf    = alloc((size_t)NTOK * DIM * 2);             // 32 MB bf16 x image
  void* winb   = alloc((size_t)EXPERTS * DIM * STATE * 2);
  void* wsinb  = alloc((size_t)EXPERTS * DIM * SELH * 2);
  void* wsoutb = alloc((size_t)EXPERTS * SELH * 4 * STATE * 2);
  void* woutb  = alloc((size_t)EXPERTS * STATE * DIM * 2);
  float* a_g    = (float*)alloc((size_t)NTOK * STATE * sizeof(float));
  float* bu_g   = (float*)alloc((size_t)NTOK * STATE * sizeof(float));
  float* c_g    = (float*)alloc((size_t)NTOK * STATE * sizeof(float));
  float* skip_g = (float*)alloc((size_t)NTOK * STATE * sizeof(float));
  float* hseq   = (float*)alloc((size_t)NTOK * STATE * sizeof(float));

  k_meta_init<<<1, 64, 0, stream>>>(meta);
  k_bucket_fill<<<(BUCKET_LEN + 255) / 256, 256, 0, stream>>>(bucket, BUCKET_LEN);
  k_routes<<<NTOK / 256, 256, 0, stream>>>(tok, routes, meta, NTOK);
  k_offsets<<<1, 1, 0, stream>>>(meta);
  k_scatter<<<NTOK / 256, 256, 0, stream>>>(routes, meta, bucket, NTOK);

  const int nx    = NTOK * DIM;                  // 16,777,216
  const int nwin  = EXPERTS * DIM * STATE;       // 1,048,576
  const int nwsin = EXPERTS * DIM * SELH;        // 2,097,152
  const int nwso  = EXPERTS * SELH * 4 * STATE;  // 1,048,576
  const int nwout = EXPERTS * STATE * DIM;       // 1,048,576
  k_cvt_x<<<(nx + 255) / 256, 256, 0, stream>>>(x, xbf, nx);
  k_pack_w<<<(nwin  + 255) / 256, 256, 0, stream>>>(W_in,      winb,   DIM,  STATE);
  k_pack_w<<<(nwsin + 255) / 256, 256, 0, stream>>>(W_sel_in,  wsinb,  DIM,  SELH);
  k_pack_w<<<(nwso  + 255) / 256, 256, 0, stream>>>(W_sel_out, wsoutb, SELH, 4 * STATE);
  k_pack_w<<<(nwout + 255) / 256, 256, 0, stream>>>(W_out,     woutb,  STATE, DIM);

  k_phase1<<<NTILES_MAX, 128, 0, stream>>>(xbf, winb, wsinb, wsoutb, d_param,
                                           bucket, meta, a_g, bu_g, c_g, skip_g);
  k_scan<<<NB_, 128, 0, stream>>>(routes, a_g, bu_g, hseq);
  k_phase2<<<NTILES_MAX, 128, 0, stream>>>(c_g, skip_g, hseq, woutb,
                                           bucket, meta, out);
}
